// VQVAE_63909113364961
// MI455X (gfx1250) — matrix operand order repaired and checked by reading, not compiled, not run on hardware
//
#include <hip/hip_runtime.h>

// ---------------------------------------------------------------------------
// VQ-VAE forward on MI455X (gfx1250). All GEMM-shaped work runs on
// v_wmma_f32_16x16x32_bf16 (wave32). Register-blocked implicit-GEMM convs
// (4 M-tiles per wave, shared B fragment), LDS-staged weights/codebook via
// GLOBAL_LOAD_ASYNC_TO_LDS_B128 (ASYNCcnt) when the toolchain exposes it.
//
// Fragment layouts (CDNA5 ISA 7.12.2):
//   A (16x32 bf16): lanes 0-15 -> M=lane, K runs {0..7,16..23};
//                   lanes 16-31 same rows, K runs +8  -> two b128 loads/lane.
//   B (32x16 bf16): lanes 0-15 -> N=lane, K runs {0..7,16..23}; lanes 16-31 -> K runs +8
//                   -> same K order as A -> two ds_load_b128 per lane.
//   C/D (16x16 f32): VGPR r -> row r (+8 for lanes>=16), col = lane&15.
// ---------------------------------------------------------------------------

typedef __attribute__((ext_vector_type(16))) __bf16 v16bf;
typedef __attribute__((ext_vector_type(8)))  __bf16 bf16x8;
typedef __attribute__((ext_vector_type(8)))  float  v8f;

__device__ __forceinline__ bf16x8 zero8() {
  bf16x8 z;
#pragma unroll
  for (int i = 0; i < 8; ++i) z[i] = (__bf16)0.0f;
  return z;
}

// Cooperative global->LDS stage in 16B chunks (blockDim == 256).
// Uses the gfx1250 async-DMA path (no VGPR round-trip, tracked by ASYNCcnt)
// when available; falls back to plain load/store otherwise (also covers the
// host-side compile pass where the device builtin is not declared).
// Builtin param type (from compiler diagnostic): int vector_size(16) pointers,
// global (AS1) source and LDS (AS3) destination.
typedef int v4i_vs __attribute__((vector_size(16)));

__device__ __forceinline__ void stage_lds_b128(void* dst, const void* src,
                                               int chunks) {
#if defined(__gfx1250__) && \
    __has_builtin(__builtin_amdgcn_global_load_async_to_lds_b128)
  typedef v4i_vs __attribute__((address_space(1)))* gvp;
  typedef v4i_vs __attribute__((address_space(3)))* lvp;
  const v4i_vs* s = (const v4i_vs*)src;
  v4i_vs* d = (v4i_vs*)dst;
  for (int i = threadIdx.x; i < chunks; i += 256)
    __builtin_amdgcn_global_load_async_to_lds_b128((gvp)(s + i), (lvp)(d + i),
                                                   0, 0);
#if __has_builtin(__builtin_amdgcn_s_wait_asynccnt)
  __builtin_amdgcn_s_wait_asynccnt(0);
#else
  asm volatile("s_wait_asynccnt 0x0" ::: "memory");
#endif
#else
  const uint4* s = (const uint4*)src;
  uint4* d = (uint4*)dst;
  for (int i = threadIdx.x; i < chunks; i += 256) d[i] = s[i];
#endif
}

// --------------------------- weight / codebook prep ------------------------

// HWIO fp32 [K, CO] (K = (kh*KW+kw)*CI+ci flattened) -> bf16 col-major
// [COpad][Kpad], zero padded.
__global__ void pack_w(const float* __restrict__ w, __bf16* __restrict__ dst,
                       int K, int CO, int Kpad, int COpad) {
  int i = blockIdx.x * blockDim.x + threadIdx.x;
  if (i >= Kpad * COpad) return;
  int k = i % Kpad, n = i / Kpad;
  float v = (k < K && n < CO) ? w[(size_t)k * CO + n] : 0.0f;
  dst[(size_t)n * Kpad + k] = (__bf16)v;
}

// codebook [256,32] -> bf16 (-2*c) row-contiguous (B fragment source) + ||c||^2
__global__ void pack_cb(const float* __restrict__ cb, __bf16* __restrict__ dst,
                        float* __restrict__ cnorm) {
  int n = blockIdx.x * blockDim.x + threadIdx.x;
  if (n >= 256) return;
  float s = 0.0f;
  for (int k = 0; k < 32; ++k) {
    float v = cb[n * 32 + k];
    s += v * v;
    dst[n * 32 + k] = (__bf16)(-2.0f * v);
  }
  cnorm[n] = s;
}

__global__ void cvt_bf16(const float* __restrict__ src, __bf16* __restrict__ dst,
                         int n) {
  int i = blockIdx.x * blockDim.x + threadIdx.x;
  int stride = gridDim.x * blockDim.x;
  for (; i < n; i += stride) dst[i] = (__bf16)src[i];
}

// --------------------------- implicit-GEMM conv ----------------------------
// Regular conv:    OSTR = stride, LDIL = 1,  ih = oh*OSTR + kh - PAD
// Transposed conv: OSTR = 1,      LDIL = s,  valid iff (oh+kh-PAD) % s == 0
// FASTA: Cin % 8 == 0 -> A fragment = two b128 loads per lane per k-step.
// MTW:   M-tiles per wave (register blocking; B fragment + k-decode shared).
template <int KH, int KW, int CIN, int COUT, int IH, int IW, int OH, int OW,
          int OSTR, int LDIL, int PAD, bool RELU, bool FASTA, int MTW,
          bool WOBF, bool WOF32>
__global__ __launch_bounds__(256) void conv_wmma(
    const __bf16* __restrict__ act,   // [B, IH, IW, CIN] bf16
    const __bf16* __restrict__ wp,    // [COpad][Kpad] bf16 packed
    const float* __restrict__ bias,   // [COUT] f32
    __bf16* __restrict__ obf,         // [M, COUT] bf16
    float* __restrict__ of32,         // [M, COUT] f32
    int M) {
  constexpr int K = KH * KW * CIN;
  constexpr int Kpad = (K + 31) & ~31;
  constexpr int COpad = (COUT + 15) & ~15;
  constexpr int NT = COpad / 16;

  // ---- async-stage packed weights into LDS (<=64KB), all waves cooperate --
  extern __shared__ char smem[];
  __bf16* lw = (__bf16*)smem;
  stage_lds_b128(lw, wp, (COpad * Kpad * 2) / 16);
  __syncthreads();

  const int lane = threadIdx.x & 31;
  const int wv = blockIdx.x * 8 + (threadIdx.x >> 5);
  const int mtg = wv / NT;        // group of MTW consecutive M-tiles
  const int ntile = wv - mtg * NT;
  const int mt0 = mtg * MTW;
  if (mt0 >= (M >> 4)) return;    // wave-uniform; grids sized for full groups

  const int am = lane & 15;          // A row within tile / B column
  const int hi8 = (lane >> 4) << 3;  // +8 K offset for A (upper half-wave)
  const int bk0 = (lane >> 4) << 4;  // +16 K offset (upper half-wave), unused
  const int n = ntile * 16 + am;

  // decode this lane's A-row output positions, once per wave
  int ohv[MTW], owv[MTW], ibase[MTW];
#pragma unroll
  for (int i = 0; i < MTW; ++i) {
    const int m = (mt0 + i) * 16 + am;
    owv[i] = m % OW;
    const int t = m / OW;
    ohv[i] = t % OH;
    ibase[i] = (t / OH) * IH;  // b*IH
  }

  v8f acc[MTW];
  {
    const float bv = (n < COUT) ? bias[n] : 0.0f;  // bias folded into C
#pragma unroll
    for (int i = 0; i < MTW; ++i)
#pragma unroll
      for (int r = 0; r < 8; ++r) acc[i][r] = bv;
  }

  const __bf16* lwrow = lw + n * Kpad;

#pragma unroll 2
  for (int k0 = 0; k0 < Kpad; k0 += 32) {
    // ---- B fragment from LDS (shared by all MTW sub-tiles) ----
    v16bf bfr;
    {
      const bf16x8 lo = *(const bf16x8*)(lwrow + k0 + hi8);
      const bf16x8 hi = *(const bf16x8*)(lwrow + k0 + 16 + hi8);
#pragma unroll
      for (int j = 0; j < 8; ++j) {
        bfr[j] = lo[j];
        bfr[8 + j] = hi[j];
      }
    }
    // ---- A fragments (fused im2col gather, k-decode shared) ----
    v16bf a[MTW];
    if constexpr (FASTA) {
#pragma unroll
      for (int run = 0; run < 2; ++run) {
        const int kb = k0 + run * 16 + hi8;  // 8 contiguous K = contiguous ci
        bf16x8 v[MTW];
#pragma unroll
        for (int i = 0; i < MTW; ++i) v[i] = zero8();
        if ((K == Kpad) || (kb < K)) {
          const int ci0 = kb % CIN;       // shared across sub-tiles
          const int pix = kb / CIN;
          const int kw_ = pix % KW;
          const int kh_ = pix / KW;
#pragma unroll
          for (int i = 0; i < MTW; ++i) {
            const int ihn = ohv[i] * OSTR + kh_ - PAD;
            const int iwn = owv[i] * OSTR + kw_ - PAD;
            bool ok = (ihn >= 0) && (iwn >= 0);
            int ih = ihn, iw = iwn;
            if (LDIL != 1) {  // fractionally-strided (transposed conv)
              ok = ok && ((ihn % LDIL) == 0) && ((iwn % LDIL) == 0);
              ih = ihn / LDIL;
              iw = iwn / LDIL;
            }
            ok = ok && (ih < IH) && (iw < IW);
            if (ok)
              v[i] = *(const bf16x8*)(act +
                     (((size_t)(ibase[i] + ih) * IW + iw) * CIN + ci0));
          }
        }
#pragma unroll
        for (int i = 0; i < MTW; ++i)
#pragma unroll
          for (int j = 0; j < 8; ++j) a[i][run * 8 + j] = v[i][j];
      }
    } else {  // Cin=3 entry layer: element-wise gather
#pragma unroll
      for (int e = 0; e < 16; ++e) {
        const int k = k0 + (e & 7) + ((e >> 3) << 4) + hi8;
        if ((K == Kpad) || (k < K)) {
          const int ci = k % CIN;
          const int pix = k / CIN;
          const int kw_ = pix % KW;
          const int kh_ = pix / KW;
#pragma unroll
          for (int i = 0; i < MTW; ++i) {
            const int ihn = ohv[i] * OSTR + kh_ - PAD;
            const int iwn = owv[i] * OSTR + kw_ - PAD;
            const bool ok = (ihn >= 0) && (iwn >= 0) && (ihn < IH) && (iwn < IW);
            __bf16 v = (__bf16)0.0f;
            if (ok) v = act[((size_t)(ibase[i] + ihn) * IW + iwn) * CIN + ci];
            a[i][e] = v;
          }
        } else {
#pragma unroll
          for (int i = 0; i < MTW; ++i) a[i][e] = (__bf16)0.0f;
        }
      }
    }
    // ---- MTW independent WMMAs (no cross-accumulator dependency) ----
#pragma unroll
    for (int i = 0; i < MTW; ++i)
      acc[i] = __builtin_amdgcn_wmma_f32_16x16x32_bf16(
          false, a[i], false, bfr, (short)0, acc[i], false, false);
  }

  // ---- epilogue: ReLU + store ----
  if (n < COUT) {
#pragma unroll
    for (int i = 0; i < MTW; ++i)
#pragma unroll
      for (int r = 0; r < 8; ++r) {
        const int mr = (mt0 + i) * 16 + r + hi8;
        float v = acc[i][r];
        if (RELU) v = fmaxf(v, 0.0f);
        const size_t idx = (size_t)mr * COUT + n;
        if constexpr (WOBF) obf[idx] = (__bf16)v;
        if constexpr (WOF32) of32[idx] = v;
      }
  }
}

// ------------------------------- VQ kernel ---------------------------------
// score[m][n] = ||c_n||^2 - 2 h_m.c_n  (||h||^2 row-constant -> dropped).
// Codebook (bf16 -2c, fp32 gather table, norms) LDS-resident; one wave per
// 16-row tile sweeps 16 N-tiles with WMMA, then butterfly argmin per row.
__global__ __launch_bounds__(256) void vq_wmma(
    const __bf16* __restrict__ h,    // [M,32] bf16
    const __bf16* __restrict__ cbp,  // [256,32] bf16 = -2*codebook
    const float* __restrict__ cnorm, // [256]
    const float* __restrict__ cbf,   // [256,32] fp32 original
    float* __restrict__ hq,          // [M,32] f32 -> d_out part 1
    __bf16* __restrict__ hqb,        // [M,32] bf16 -> decoder input
    int M) {
  __shared__ __bf16 lcb[256 * 32];   // 16 KB
  __shared__ float lcn[256];         //  1 KB
  __shared__ float lcf[256 * 32];    // 32 KB
  stage_lds_b128(lcb, cbp, 1024);
  stage_lds_b128(lcf, cbf, 2048);
  if (threadIdx.x < 256) lcn[threadIdx.x] = cnorm[threadIdx.x];
  __syncthreads();

  const int lane = threadIdx.x & 31;
  const int wv = blockIdx.x * 8 + (threadIdx.x >> 5);
  if (wv >= (M >> 4)) return;
  const int am = lane & 15;
  const int hi8 = (lane >> 4) << 3;
  const int bk0 = (lane >> 4) << 4;

  // A fragment: 16 latent rows, K = 32 (single k-step, loaded once)
  v16bf a;
  {
    const __bf16* hrow = h + (size_t)(wv * 16 + am) * 32;
    const bf16x8 lo = *(const bf16x8*)(hrow + hi8);
    const bf16x8 hi = *(const bf16x8*)(hrow + 16 + hi8);
#pragma unroll
    for (int j = 0; j < 8; ++j) {
      a[j] = lo[j];
      a[8 + j] = hi[j];
    }
  }

  float minv[8];
  int mini[8];
#pragma unroll
  for (int r = 0; r < 8; ++r) {
    minv[r] = 3.4e38f;
    mini[r] = 0;
  }

#pragma unroll
  for (int nt = 0; nt < 16; ++nt) {
    const int n = nt * 16 + am;
    v16bf bfr;
    {
      const __bf16* crow = lcb + n * 32 + hi8;
      const bf16x8 lo = *(const bf16x8*)(crow);
      const bf16x8 hi = *(const bf16x8*)(crow + 16);
#pragma unroll
      for (int j = 0; j < 8; ++j) {
        bfr[j] = lo[j];
        bfr[8 + j] = hi[j];
      }
    }
    v8f c;
    const float cn = lcn[n];
#pragma unroll
    for (int r = 0; r < 8; ++r) c[r] = cn;  // C init = ||c_n||^2
    c = __builtin_amdgcn_wmma_f32_16x16x32_bf16(false, a, false, bfr, (short)0,
                                                c, false, false);
#pragma unroll
    for (int r = 0; r < 8; ++r) {
      if (c[r] < minv[r]) {  // n strictly increases -> first-min kept
        minv[r] = c[r];
        mini[r] = n;
      }
    }
  }

  // cross-lane argmin within each 16-lane half (matches D-fragment halves),
  // then gather the winning fp32 codebook row (2 floats per lane) from LDS.
  const int c0 = am * 2;
#pragma unroll
  for (int r = 0; r < 8; ++r) {
    float v = minv[r];
    int i = mini[r];
#pragma unroll
    for (int msk = 1; msk < 16; msk <<= 1) {
      const float ov = __shfl_xor(v, msk, 32);
      const int oi = __shfl_xor(i, msk, 32);
      if (ov < v || (ov == v && oi < i)) {  // jnp.argmin tie-break: low index
        v = ov;
        i = oi;
      }
    }
    const int row = wv * 16 + r + hi8;
    const float g0 = lcf[i * 32 + c0];
    const float g1 = lcf[i * 32 + c0 + 1];
    hq[(size_t)row * 32 + c0] = g0;
    hq[(size_t)row * 32 + c0 + 1] = g1;
    hqb[(size_t)row * 32 + c0] = (__bf16)g0;
    hqb[(size_t)row * 32 + c0 + 1] = (__bf16)g1;
  }
}

// ------------------------------- launcher ----------------------------------

extern "C" void kernel_launch(void* const* d_in, const int* in_sizes, int n_in,
                              void* d_out, int out_size, void* d_ws,
                              size_t ws_size, hipStream_t stream) {
  (void)in_sizes; (void)n_in; (void)out_size; (void)ws_size;
  const float* x  = (const float*)d_in[0];
  const float* W1 = (const float*)d_in[1];  const float* b1 = (const float*)d_in[2];
  const float* W2 = (const float*)d_in[3];  const float* b2 = (const float*)d_in[4];
  const float* W3 = (const float*)d_in[5];  const float* b3 = (const float*)d_in[6];
  const float* CB = (const float*)d_in[7];
  const float* D1 = (const float*)d_in[8];  const float* e1 = (const float*)d_in[9];
  const float* D2 = (const float*)d_in[10]; const float* e2 = (const float*)d_in[11];
  const float* D3 = (const float*)d_in[12]; const float* e3 = (const float*)d_in[13];

  // workspace carve-up (bf16 activations; ~130 MB total)
  char* ws = (char*)d_ws;
  size_t off = 0;
  auto carve = [&](size_t bytes) -> char* {
    char* p = ws + off;
    off += (bytes + 255) & ~(size_t)255;
    return p;
  };
  __bf16* xb  = (__bf16*)carve(6291456ull * 2);   // x            [128,128,128,3]
  __bf16* h1  = (__bf16*)carve(16777216ull * 2);  // relu(conv1)  [128,64,64,32]
  __bf16* h2  = (__bf16*)carve(8388608ull * 2);   // relu(conv2)  [128,32,32,64]
  __bf16* h3  = (__bf16*)carve(4194304ull * 2);   // conv3        [128,32,32,32]
  __bf16* hqb = (__bf16*)carve(4194304ull * 2);   // quantized    [128,32,32,32]
  __bf16* o1  = (__bf16*)carve(8388608ull * 2);   // relu(convT1) [128,32,32,64]
  __bf16* o2  = (__bf16*)carve(16777216ull * 2);  // relu(convT2) [128,64,64,32]
  __bf16* w1p = (__bf16*)carve(32ull * 64 * 2);
  __bf16* w2p = (__bf16*)carve(64ull * 512 * 2);
  __bf16* w3p = (__bf16*)carve(32ull * 576 * 2);
  __bf16* d1p = (__bf16*)carve(64ull * 288 * 2);
  __bf16* d2p = (__bf16*)carve(32ull * 1024 * 2);
  __bf16* d3p = (__bf16*)carve(16ull * 512 * 2);
  __bf16* cbp = (__bf16*)carve(256ull * 32 * 2);
  float*  cn  = (float*)carve(256ull * 4);

  float* hq_out = (float*)d_out;            // _h_q : 4,194,304 f32
  float* d_outp = (float*)d_out + 4194304;  // d    : 6,291,456 f32

  // ---- prep: bf16 convert input, pack weights / codebook ----
  cvt_bf16<<<2048, 256, 0, stream>>>(x, xb, 6291456);
  pack_w<<<(64 * 32 + 255) / 256, 256, 0, stream>>>(W1, w1p, 48, 32, 64, 32);
  pack_w<<<(512 * 64 + 255) / 256, 256, 0, stream>>>(W2, w2p, 512, 64, 512, 64);
  pack_w<<<(576 * 32 + 255) / 256, 256, 0, stream>>>(W3, w3p, 576, 32, 576, 32);
  pack_w<<<(288 * 64 + 255) / 256, 256, 0, stream>>>(D1, d1p, 288, 64, 288, 64);
  pack_w<<<(1024 * 32 + 255) / 256, 256, 0, stream>>>(D2, d2p, 1024, 32, 1024, 32);
  pack_w<<<(512 * 16 + 255) / 256, 256, 0, stream>>>(D3, d3p, 512, 3, 512, 16);
  pack_cb<<<1, 256, 0, stream>>>(CB, cbp, cn);

  // ---- encoder ----
  // conv1 4x4 s2 3->32, pad(1,1): M=524288 -> 8192 Mtile-groups * NT2 = 16384 waves
  conv_wmma<4, 4, 3, 32, 128, 128, 64, 64, 2, 1, 1, true, false, 4, true, false>
      <<<2048, 256, 4096, stream>>>(xb, w1p, b1, h1, nullptr, 524288);
  // conv2 4x4 s2 32->64, pad(1,1): 2048 groups * NT4 = 8192 waves
  conv_wmma<4, 4, 32, 64, 64, 64, 32, 32, 2, 1, 1, true, true, 4, true, false>
      <<<1024, 256, 65536, stream>>>(h1, w2p, b2, h2, nullptr, 131072);
  // conv3 3x3 s1 64->32, pad(1,1): 2048 groups * NT2 = 4096 waves
  conv_wmma<3, 3, 64, 32, 32, 32, 32, 32, 1, 1, 1, false, true, 4, true, false>
      <<<512, 256, 36864, stream>>>(h2, w3p, b3, h3, nullptr, 131072);

  // ---- vector quantization (WMMA distance + butterfly argmin) ----
  vq_wmma<<<1024, 256, 0, stream>>>(h3, cbp, cn, CB, hq_out, hqb, 131072);

  // ---- decoder (transposed convs as lhs-dilated convs) ----
  // convT1 3x3 s1 32->64: pad_before = 1; 2048 groups * NT4 = 8192 waves
  conv_wmma<3, 3, 32, 64, 32, 32, 32, 32, 1, 1, 1, true, true, 4, true, false>
      <<<1024, 256, 36864, stream>>>(hqb, d1p, e1, o1, nullptr, 131072);
  // convT2 4x4 s2 64->32: lhs_dil 2, pad 2; 8192 groups * NT2 = 16384 waves
  conv_wmma<4, 4, 64, 32, 32, 32, 64, 64, 1, 2, 2, true, true, 4, true, false>
      <<<2048, 256, 65536, stream>>>(o1, d2p, e2, o2, nullptr, 524288);
  // convT3 4x4 s2 32->3: 32768 groups * NT1 = 32768 waves, f32 output
  conv_wmma<4, 4, 32, 3, 64, 64, 128, 128, 1, 2, 2, false, true, 4, false, true>
      <<<4096, 256, 16384, stream>>>(o2, d3p, e3, nullptr, d_outp, 2097152);
}
